// InceptionDWConv2d_5360119185887
// MI455X (gfx1250) — matrix operand order repaired and checked by reading, not compiled, not run on hardware
//
#include <hip/hip_runtime.h>
#include <hip/hip_bf16.h>

typedef __attribute__((ext_vector_type(16))) _Float16 v16h;
typedef __attribute__((ext_vector_type(8)))  _Float16 v8h;
typedef __attribute__((ext_vector_type(8)))  float    v8f;

// ---------------------------------------------------------------------------
// WMMA fragment helpers (CDNA5 16x16x32 f16, wave32).
// A: 16x32 (MxK). lane&15 = row M; lane>>4 selects K-half (kb = 0 or 8).
//    elements 0..7  -> K = kb+0..kb+7   (contiguous 16B)
//    elements 8..15 -> K = kb+16..kb+23 (contiguous 16B)
// B: 32x16 (KxN). lane&15 = col N; lane>>4 selects K-half; element e -> K=kb+e (e<8), kb+8+e (e>=8).
//    Weights are pre-swizzled fragment-major, so B load = 32 contiguous bytes.
// C/D: 16x16 f32. VGPR r -> row = r + 8*(lane>>4), col = lane&15.
// ---------------------------------------------------------------------------
static __device__ __forceinline__ v16h load_a_rm(const _Float16* p, int ld) {
  const int lane = threadIdx.x & 31;
  const int row  = lane & 15;
  const int kb   = (lane >> 4) * 8;
  const _Float16* r = p + (size_t)row * ld + kb;
  v8h lo = *(const v8h*)(r);
  v8h hi = *(const v8h*)(r + 16);
  return __builtin_shufflevector(lo, hi, 0, 1, 2, 3, 4, 5, 6, 7,
                                 8, 9, 10, 11, 12, 13, 14, 15);
}

// base = start of fragment-major weight region; tile = kt * Ntiles + nt
static __device__ __forceinline__ v16h load_b_frag(const _Float16* base,
                                                   int tile) {
  const int lane = threadIdx.x & 31;
  return *(const v16h*)(base + (((size_t)tile * 32 + lane) << 4));
}

static __device__ __forceinline__ v8f wmma16(v16h a, v16h b, v8f c) {
  return __builtin_amdgcn_wmma_f32_16x16x32_f16(false, a, false, b,
                                                (short)0, c, false, false);
}

// ---------------------------------------------------------------------------
// K0: convert all weights to f16 in fragment-major (per 32x16 tile, per-lane
//     contiguous 16-half fragments). Region element offsets (f16):
//   [0,8192)        expand_w  K=64  N=128 (Ntiles=8)
//   [8192,16384)    concat_w  K=128 N=64  (Ntiles=4)
//   [16384,32768)   in_w      K=64  N=256 (Ntiles=16)
//   [32768,38912)   xproj_w   K=128 N=48 padded from 36 (Ntiles=3)
//   [38912,47104)   out_w     K=128 N=64  (Ntiles=4)
//   [47104,83968)   conv_w: 9 taps x (K=64,N=64) from OIHW
// ---------------------------------------------------------------------------
__global__ void k_prep(const float* __restrict__ expand_w,
                       const float* __restrict__ concat_w,
                       const float* __restrict__ in_w,
                       const float* __restrict__ xproj_w,
                       const float* __restrict__ out_w,
                       const float* __restrict__ conv_w,
                       _Float16* __restrict__ wf) {
  int idx = blockIdx.x * blockDim.x + threadIdx.x;
  if (idx >= 83968) return;
  int j, Ntiles, region;
  if (idx < 8192)       { j = idx;         Ntiles = 8;  region = 0; }
  else if (idx < 16384) { j = idx - 8192;  Ntiles = 4;  region = 1; }
  else if (idx < 32768) { j = idx - 16384; Ntiles = 16; region = 2; }
  else if (idx < 38912) { j = idx - 32768; Ntiles = 3;  region = 3; }
  else if (idx < 47104) { j = idx - 38912; Ntiles = 4;  region = 4; }
  else                  { j = idx - 47104; Ntiles = 4;  region = 5; }
  int e    = j & 15;
  int lane = (j >> 4) & 31;
  int tile = j >> 9;
  int kt, nt, tap = 0;
  if (region == 5) {           // conv: 8 tiles per tap (Ktiles=2 x Ntiles=4)
    tap = tile >> 3;
    int tt = tile & 7;
    kt = tt >> 2;
    nt = tt & 3;
  } else {
    kt = tile / Ntiles;
    nt = tile - kt * Ntiles;
  }
  int k   = kt * 32 + (lane >> 4) * 8 + (e & 7) + (e >> 3) * 16;
  int col = nt * 16 + (lane & 15);
  float v;
  switch (region) {
    case 0:  v = expand_w[k * 128 + col]; break;
    case 1:  v = concat_w[k * 64 + col];  break;
    case 2:  v = in_w[k * 256 + col];     break;
    case 3:  v = (col < 36) ? xproj_w[k * 36 + col] : 0.f; break;
    case 4:  v = out_w[k * 64 + col];     break;
    default: {
      int ky = tap / 3, kx = tap - ky * 3;   // B[k=ci][col=o] from OIHW
      v = conv_w[((col * 64 + k) * 3 + ky) * 3 + kx];
      break;
    }
  }
  wf[idx] = (_Float16)v;
}

// ---------------------------------------------------------------------------
// K1: front MLP per 16-token tile: x_id -> expand -> concat(+b) -> t,
//     layernorm -> in-proj(+b) -> xs, z.  One wave per tile.
// ---------------------------------------------------------------------------
__global__ __launch_bounds__(32) void k_front(
    const float* __restrict__ x,
    const _Float16* __restrict__ w_expand, const _Float16* __restrict__ w_concat,
    const float* __restrict__ concat_b, const float* __restrict__ ln_g,
    const float* __restrict__ ln_b, const _Float16* __restrict__ w_in,
    const float* __restrict__ in_b,
    float* __restrict__ t_buf, float* __restrict__ xs_buf,
    float* __restrict__ z_buf) {
  __shared__ __align__(16) _Float16 sA[16 * 64];
  __shared__ __align__(16) _Float16 sT1[16 * 128];
  __shared__ __align__(16) float    sT[16 * 64];
  __shared__ __align__(16) _Float16 sTn[16 * 64];

  const int lane   = threadIdx.x;
  const int token0 = blockIdx.x * 16;
  const int b      = token0 >> 12;       // 4096 tokens per batch
  const int hw0    = token0 & 4095;
  const float* xb  = x + (size_t)b * 128 * 4096;

  // stage x_id tile (16 tokens x 64 channels), transpose NCHW -> token-major
  for (int i = lane; i < 16 * 64; i += 32) {
    int c = i >> 4, row = i & 15;
    sA[row * 64 + c] = (_Float16)xb[(size_t)c * 4096 + hw0 + row];
  }
  __syncthreads();

  // t1 = A(16x64) @ expand(64x128)
  for (int nt = 0; nt < 8; ++nt) {
    v8f acc = {};
    for (int kt = 0; kt < 2; ++kt) {
      v16h a  = load_a_rm(sA + kt * 32, 64);
      v16h bb = load_b_frag(w_expand, kt * 8 + nt);
      acc = wmma16(a, bb, acc);
    }
    int col = lane & 15, rb = (lane >> 4) * 8;
#pragma unroll
    for (int r = 0; r < 8; ++r)
      sT1[(rb + r) * 128 + nt * 16 + col] = (_Float16)acc[r];
  }
  __syncthreads();

  // t = t1(16x128) @ concat(128x64) + concat_b
  for (int nt = 0; nt < 4; ++nt) {
    v8f acc = {};
    for (int kt = 0; kt < 4; ++kt) {
      v16h a  = load_a_rm(sT1 + kt * 32, 128);
      v16h bb = load_b_frag(w_concat, kt * 4 + nt);
      acc = wmma16(a, bb, acc);
    }
    int col = lane & 15, rb = (lane >> 4) * 8;
#pragma unroll
    for (int r = 0; r < 8; ++r)
      sT[(rb + r) * 64 + nt * 16 + col] = acc[r] + concat_b[nt * 16 + col];
  }
  __syncthreads();

  // spill t for the residual path (token-major f32)
  for (int i = lane; i < 16 * 64; i += 32) {
    int row = i >> 6, c = i & 63;
    t_buf[(size_t)(token0 + row) * 64 + c] = sT[row * 64 + c];
  }

  // layernorm over 64 features: 2 lanes per row, combine across halves
  {
    int row = lane & 15, half = lane >> 4;
    float s = 0.f, s2 = 0.f;
    for (int c = half * 32; c < half * 32 + 32; ++c) {
      float v = sT[row * 64 + c];
      s += v; s2 += v * v;
    }
    s  += __shfl_xor(s, 16);
    s2 += __shfl_xor(s2, 16);
    float mean = s * (1.f / 64.f);
    float var  = s2 * (1.f / 64.f) - mean * mean;
    float rs   = rsqrtf(var + 1e-5f);
    for (int c = half * 32; c < half * 32 + 32; ++c) {
      float v = (sT[row * 64 + c] - mean) * rs * ln_g[c] + ln_b[c];
      sTn[row * 64 + c] = (_Float16)v;
    }
  }
  __syncthreads();

  // xz = tn(16x64) @ in_w(64x256) + in_b -> xs (cols 0..127), z (128..255)
  for (int nt = 0; nt < 16; ++nt) {
    v8f acc = {};
    for (int kt = 0; kt < 2; ++kt) {
      v16h a  = load_a_rm(sTn + kt * 32, 64);
      v16h bb = load_b_frag(w_in, kt * 16 + nt);
      acc = wmma16(a, bb, acc);
    }
    int col = lane & 15, rb = (lane >> 4) * 8;
    int n = nt * 16 + col;
    float bias = in_b[n];
    float* dst = (nt < 8) ? xs_buf : z_buf;
    int nn = (nt < 8) ? n : (n - 128);
#pragma unroll
    for (int r = 0; r < 8; ++r)
      dst[(size_t)(token0 + rb + r) * 128 + nn] = acc[r] + bias;
  }
}

// ---------------------------------------------------------------------------
// K2: causal depthwise conv1d (K=3) + SiLU, elementwise.
// ---------------------------------------------------------------------------
__global__ void k_conv1d(const float* __restrict__ xs,
                         const float* __restrict__ w1d,
                         const float* __restrict__ b1d,
                         float* __restrict__ xcf, _Float16* __restrict__ xch,
                         int total) {
  int i = blockIdx.x * blockDim.x + threadIdx.x;
  if (i >= total) return;
  int d = i & 127;
  int l = (i >> 7) & 4095;
  float acc = b1d[d];
#pragma unroll
  for (int k = 0; k < 3; ++k) {
    int ll = l + k - 2;
    if (ll >= 0) acc += xs[i + (k - 2) * 128] * w1d[k * 128 + d];
  }
  float s = acc / (1.f + __expf(-acc));  // silu
  xcf[i] = s;
  xch[i] = (_Float16)s;
}

// ---------------------------------------------------------------------------
// K3: dbc = xc(16x128) @ xproj(128x48 padded); dt = softplus(dbc[:, :4]@dt_w
//     + dt_b); extract B (cols 4..19) and C (cols 20..35).
// ---------------------------------------------------------------------------
__global__ __launch_bounds__(32) void k_xproj(
    const _Float16* __restrict__ xch, const _Float16* __restrict__ wxp,
    const float* __restrict__ dt_w, const float* __restrict__ dt_b,
    float* __restrict__ dt_buf, float* __restrict__ Bm, float* __restrict__ Cm) {
  __shared__ float sD[16 * 48];
  const int lane   = threadIdx.x;
  const int token0 = blockIdx.x * 16;

  for (int nt = 0; nt < 3; ++nt) {
    v8f acc = {};
    for (int kt = 0; kt < 4; ++kt) {
      v16h a  = load_a_rm(xch + (size_t)token0 * 128 + kt * 32, 128);
      v16h bb = load_b_frag(wxp, kt * 3 + nt);
      acc = wmma16(a, bb, acc);
    }
    int col = lane & 15, rb = (lane >> 4) * 8;
#pragma unroll
    for (int r = 0; r < 8; ++r)
      sD[(rb + r) * 48 + nt * 16 + col] = acc[r];
  }
  __syncthreads();

  for (int i = lane; i < 16 * 128; i += 32) {
    int row = i >> 7, d = i & 127;
    float acc = dt_b[d];
#pragma unroll
    for (int r = 0; r < 4; ++r) acc += sD[row * 48 + r] * dt_w[r * 128 + d];
    float sp = (acc > 20.f) ? acc : log1pf(__expf(acc));
    dt_buf[(size_t)(token0 + row) * 128 + d] = sp;
  }
  for (int i = lane; i < 16 * 16; i += 32) {
    int row = i >> 4, n = i & 15;
    Bm[(size_t)(token0 + row) * 16 + n] = sD[row * 48 + 4 + n];
    Cm[(size_t)(token0 + row) * 16 + n] = sD[row * 48 + 20 + n];
  }
}

// ---------------------------------------------------------------------------
// K4: SSM scan. 16 lanes per (b,d), one state n per lane; shfl_xor tree for
//     the C-contraction; fused +xc*D and *silu(z) gate; emits g in f16.
// ---------------------------------------------------------------------------
__global__ __launch_bounds__(256) void k_scan(
    const float* __restrict__ dt_buf, const float* __restrict__ xcf,
    const float* __restrict__ Bm, const float* __restrict__ Cm,
    const float* __restrict__ A_log, const float* __restrict__ D_ssm,
    const float* __restrict__ z_buf, _Float16* __restrict__ gh) {
  int t = blockIdx.x * 256 + threadIdx.x;
  int p = t >> 4;          // (b,d) pair, 0..1023
  int n = t & 15;
  int b = p >> 7;
  int d = p & 127;
  float Aneg = -__expf(A_log[d * 16 + n]);
  float Dv   = D_ssm[d];
  float h = 0.f;
  const size_t base0 = (size_t)b * 4096;
  for (int l = 0; l < 4096; ++l) {
    size_t tok = base0 + l;
    float dt = dt_buf[tok * 128 + d];
    float xc = xcf[tok * 128 + d];
    float Bn = Bm[tok * 16 + n];
    float Cn = Cm[tok * 16 + n];
    float dA = __expf(dt * Aneg);
    h = fmaf(dA, h, dt * Bn * xc);
    float prod = h * Cn;
    prod += __shfl_xor(prod, 1);
    prod += __shfl_xor(prod, 2);
    prod += __shfl_xor(prod, 4);
    prod += __shfl_xor(prod, 8);
    if (n == 0) {
      float z = z_buf[tok * 128 + d];
      float y = prod + Dv * xc;
      float g = y * (z / (1.f + __expf(-z)));
      gh[tok * 128 + d] = (_Float16)g;
    }
  }
}

// ---------------------------------------------------------------------------
// K5: out = g(16x128) @ out_w(128x64) + out_b + t, ReLU, -> NCHW ch 0..63.
// ---------------------------------------------------------------------------
__global__ __launch_bounds__(32) void k_out(
    const _Float16* __restrict__ gh, const _Float16* __restrict__ w_out,
    const float* __restrict__ out_b, const float* __restrict__ t_buf,
    float* __restrict__ out) {
  const int lane   = threadIdx.x;
  const int token0 = blockIdx.x * 16;
  const int b      = token0 >> 12;
  const int hw0    = token0 & 4095;
  for (int nt = 0; nt < 4; ++nt) {
    v8f acc = {};
    for (int kt = 0; kt < 4; ++kt) {
      v16h a  = load_a_rm(gh + (size_t)token0 * 128 + kt * 32, 128);
      v16h bb = load_b_frag(w_out, kt * 4 + nt);
      acc = wmma16(a, bb, acc);
    }
    int col = lane & 15, rb = (lane >> 4) * 8;
    int c = nt * 16 + col;
    float bias = out_b[c];
#pragma unroll
    for (int r = 0; r < 8; ++r) {
      int row = rb + r;
      float v = acc[r] + bias + t_buf[(size_t)(token0 + row) * 64 + c];
      out[((size_t)b * 128 + c) * 4096 + hw0 + row] = fmaxf(v, 0.f);
    }
  }
}

// ---------------------------------------------------------------------------
// K6: 3x3 conv on x_hw via implicit GEMM. One wave per (b, h, 16-wide w tile).
//     LDS: 3 padded rows, row-major [w-pos][channel], pitch 72 halves so
//     A-fragments are two contiguous 16B chunks (ds_load_b128) and rows map
//     to distinct bank groups. 9 taps x 2 K-steps x 4 N-tiles = 72 WMMAs.
// ---------------------------------------------------------------------------
__global__ __launch_bounds__(32) void k_conv2d(
    const float* __restrict__ x, const _Float16* __restrict__ wc,
    const float* __restrict__ conv_b, float* __restrict__ out) {
  __shared__ __align__(16) _Float16 sX[3][18 * 72];
  const int lane = threadIdx.x;
  const int w0 = (blockIdx.x & 3) * 16;
  const int h  = (blockIdx.x >> 2) & 63;
  const int b  = blockIdx.x >> 8;
  const float* xb = x + ((size_t)b * 128 + 64) * 4096;

  for (int dy = 0; dy < 3; ++dy) {
    int hh = h + dy - 1;
    for (int i = lane; i < 64 * 18; i += 32) {
      int c = i / 18;
      int j = i - c * 18;          // w-pos 0..17 (ww = w0-1+j)
      int ww = w0 - 1 + j;
      float v = 0.f;
      if (hh >= 0 && hh < 64 && ww >= 0 && ww < 64)
        v = xb[(size_t)c * 4096 + hh * 64 + ww];
      sX[dy][j * 72 + c] = (_Float16)v;
    }
  }
  __syncthreads();

  v8f zero = {};
  v8f acc[4];
#pragma unroll
  for (int nt = 0; nt < 4; ++nt) acc[nt] = zero;

  for (int dy = 0; dy < 3; ++dy) {
    for (int dx = 0; dx < 3; ++dx) {
      const int tap = dy * 3 + dx;
      for (int kt = 0; kt < 2; ++kt) {
        // A rows = output w positions (shifted by dx), K = input channels
        v16h a = load_a_rm(sX[dy] + dx * 72 + kt * 32, 72);
#pragma unroll
        for (int nt = 0; nt < 4; ++nt) {
          v16h bb = load_b_frag(wc, tap * 8 + kt * 4 + nt);
          acc[nt] = wmma16(a, bb, acc[nt]);
        }
      }
    }
  }

  int col = lane & 15, rb = (lane >> 4) * 8;
  for (int nt = 0; nt < 4; ++nt) {
    int c = nt * 16 + col;
    float bias = conv_b[c];
#pragma unroll
    for (int r = 0; r < 8; ++r) {
      float v = fmaxf(acc[nt][r] + bias, 0.f);
      out[((size_t)b * 128 + 64 + c) * 4096 + (size_t)h * 64 + w0 + rb + r] = v;
    }
  }
}

// ---------------------------------------------------------------------------
extern "C" void kernel_launch(void* const* d_in, const int* in_sizes, int n_in,
                              void* d_out, int out_size, void* d_ws,
                              size_t ws_size, hipStream_t stream) {
  const float* x        = (const float*)d_in[0];
  const float* conv_w   = (const float*)d_in[1];
  const float* conv_b   = (const float*)d_in[2];
  const float* expand_w = (const float*)d_in[3];
  const float* concat_w = (const float*)d_in[4];
  const float* concat_b = (const float*)d_in[5];
  const float* ln_g     = (const float*)d_in[6];
  const float* ln_b     = (const float*)d_in[7];
  const float* in_w     = (const float*)d_in[8];
  const float* in_b     = (const float*)d_in[9];
  const float* conv1d_w = (const float*)d_in[10];
  const float* conv1d_b = (const float*)d_in[11];
  const float* xproj_w  = (const float*)d_in[12];
  const float* dt_w     = (const float*)d_in[13];
  const float* dt_b     = (const float*)d_in[14];
  const float* A_log    = (const float*)d_in[15];
  const float* D_ssm    = (const float*)d_in[16];
  const float* out_w    = (const float*)d_in[17];
  const float* out_b    = (const float*)d_in[18];
  float* out = (float*)d_out;
  (void)in_sizes; (void)n_in; (void)out_size; (void)ws_size;

  char* ws = (char*)d_ws;
  size_t off = 0;
  auto alloc = [&](size_t bytes) {
    char* p = ws + off;
    off = (off + bytes + 255) & ~(size_t)255;
    return p;
  };
  const size_t NTOK = 32768;  // B * H * W
  _Float16* wf  = (_Float16*)alloc(83968 * 2);
  float* t_buf  = (float*)alloc(NTOK * 64 * 4);
  float* xs_buf = (float*)alloc(NTOK * 128 * 4);
  float* z_buf  = (float*)alloc(NTOK * 128 * 4);
  float* xcf    = (float*)alloc(NTOK * 128 * 4);
  float* dt_buf = (float*)alloc(NTOK * 128 * 4);
  float* Bm     = (float*)alloc(NTOK * 16 * 4);
  float* Cm     = (float*)alloc(NTOK * 16 * 4);
  _Float16* xch = (_Float16*)alloc(NTOK * 128 * 2);
  _Float16* gh  = (_Float16*)alloc(NTOK * 128 * 2);

  const _Float16* w_expand = wf;
  const _Float16* w_concat = wf + 8192;
  const _Float16* w_in     = wf + 16384;
  const _Float16* w_xproj  = wf + 32768;
  const _Float16* w_out    = wf + 38912;
  const _Float16* w_conv   = wf + 47104;

  k_prep<<<(83968 + 255) / 256, 256, 0, stream>>>(expand_w, concat_w, in_w,
                                                  xproj_w, out_w, conv_w, wf);
  k_front<<<2048, 32, 0, stream>>>(x, w_expand, w_concat, concat_b, ln_g, ln_b,
                                   w_in, in_b, t_buf, xs_buf, z_buf);
  k_conv1d<<<4194304 / 256, 256, 0, stream>>>(xs_buf, conv1d_w, conv1d_b, xcf,
                                              xch, 4194304);
  k_xproj<<<2048, 32, 0, stream>>>(xch, w_xproj, dt_w, dt_b, dt_buf, Bm, Cm);
  k_scan<<<64, 256, 0, stream>>>(dt_buf, xcf, Bm, Cm, A_log, D_ssm, z_buf, gh);
  k_out<<<2048, 32, 0, stream>>>(gh, w_out, out_b, t_buf, out);
  k_conv2d<<<2048, 32, 0, stream>>>(x, w_conv, conv_b, out);
}
